// MultiHeadAttention_81655918231617
// MI455X (gfx1250) — matrix operand order repaired and checked by reading, not compiled, not run on hardware
//
#include <hip/hip_runtime.h>
#include <hip/hip_bf16.h>

typedef __attribute__((ext_vector_type(16))) __bf16 v16bf;
typedef __attribute__((ext_vector_type(8)))  __bf16 v8bf;
typedef __attribute__((ext_vector_type(8)))  float  v8f;

#define DM   1024
#define HEADS 16
#define DK   64
#define BATCH 2
#define SEQ  2048
#define MTOT (BATCH * SEQ)   // 4096

static __device__ inline v8f wmma_bf16(v16bf a, v16bf b, v8f c) {
  return __builtin_amdgcn_wmma_f32_16x16x32_bf16(false, a, false, b, (short)0, c,
                                                 false, false);
}

// A-fragment (16x32 bf16, rows striped in-lane) from row-major [rows, lda]
static __device__ inline v16bf load_a_frag(const __bf16* A, int lda, int row0, int k0) {
  int lane = threadIdx.x & 31;
  int r    = row0 + (lane & 15);
  int hlf  = lane >> 4;
  const __bf16* base = A + (size_t)r * lda + k0 + hlf * 16;
  v8bf lo = *(const v8bf*)(base);        // k = k0 + 16h .. +7
  v8bf hi = *(const v8bf*)(base + 8);    // k = k0 + 16h + 8 .. +15
  return __builtin_shufflevector(lo, hi, 0,1,2,3,4,5,6,7,8,9,10,11,12,13,14,15);
}

// B-fragment (32x16 bf16) from BT stored as [N rows, K cols] row-major
static __device__ inline v16bf load_bT_frag(const __bf16* BT, int ldb, int col0, int k0) {
  int lane = threadIdx.x & 31;
  int n    = col0 + (lane & 15);
  int kb   = k0 + ((lane >> 4) << 4);    // 16 contiguous k
  return *(const v16bf*)(BT + (size_t)n * ldb + kb);
}

// ---------------- prep kernels ----------------
__global__ void cast_f32_bf16_kernel(const float* __restrict__ src,
                                     __bf16* __restrict__ dst, int n) {
  int i = blockIdx.x * blockDim.x + threadIdx.x;
  if (i < n) dst[i] = (__bf16)src[i];
}

// WT[n*K + k] = (bf16) W[k*N + n]
__global__ void transpose_cast_kernel(const float* __restrict__ W,
                                      __bf16* __restrict__ WT, int K, int N) {
  __shared__ float tile[16][17];
  int k = blockIdx.x * 16 + threadIdx.y;
  int n = blockIdx.y * 16 + threadIdx.x;
  tile[threadIdx.y][threadIdx.x] = W[(size_t)k * N + n];
  __syncthreads();
  int n2 = blockIdx.y * 16 + threadIdx.y;
  int k2 = blockIdx.x * 16 + threadIdx.x;
  WT[(size_t)n2 * K + k2] = (__bf16)tile[threadIdx.x][threadIdx.y];
}

// ---------------- GEMM: C = A[M,K] * W + bias ----------------
// WT is W pre-transposed [N,K] bf16. Each wave: 16x64 tile, A-fragment reused
// across 4 WMMAs, double-buffered k-steps so loads overlap the matrix pipe.
// mode: 0 -> bf16 out [B,H,L,DK]; 1 -> bf16 out [B,H,DK,L] (V^T); 2 -> f32 [M,N]
__global__ void gemm_bf16_wmma_kernel(const __bf16* __restrict__ A,
                                      const __bf16* __restrict__ WT,
                                      const float* __restrict__ bias,
                                      void* __restrict__ out,
                                      int M, int Kdim, int Ndim, int mode) {
  int warp   = threadIdx.x >> 5;
  int tile_m = blockIdx.x * 4 + warp;          // 16 rows per wave
  int n0     = blockIdx.y * 64;                // 64 cols per wave
  int lane   = threadIdx.x & 31;
  int hlf    = lane >> 4;
  int colc   = lane & 15;
  int row0   = tile_m * 16;

  v8f acc[4] = {};
  v16bf af = load_a_frag(A, Kdim, row0, 0);
  v16bf b0 = load_bT_frag(WT, Kdim, n0,      0);
  v16bf b1 = load_bT_frag(WT, Kdim, n0 + 16, 0);
  v16bf b2 = load_bT_frag(WT, Kdim, n0 + 32, 0);
  v16bf b3 = load_bT_frag(WT, Kdim, n0 + 48, 0);

  for (int k0 = 0; k0 < Kdim; k0 += 32) {
    // next k-step (last iteration harmlessly reloads current; keeps loads
    // unconditional and clause-groupable)
    int kn = (k0 + 32 < Kdim) ? (k0 + 32) : k0;
    if (k0 + 256 < Kdim) {   // prefetch 8 k-steps ahead
      __builtin_prefetch(A + (size_t)(row0 + colc) * Kdim + k0 + 256, 0, 1);
      __builtin_prefetch(WT + (size_t)(n0 + (lane & 63)) * Kdim + k0 + 256, 0, 1);
    }
    v16bf afn = load_a_frag(A, Kdim, row0, kn);
    v16bf b0n = load_bT_frag(WT, Kdim, n0,      kn);
    v16bf b1n = load_bT_frag(WT, Kdim, n0 + 16, kn);
    v16bf b2n = load_bT_frag(WT, Kdim, n0 + 32, kn);
    v16bf b3n = load_bT_frag(WT, Kdim, n0 + 48, kn);

    acc[0] = wmma_bf16(af, b0, acc[0]);
    acc[1] = wmma_bf16(af, b1, acc[1]);
    acc[2] = wmma_bf16(af, b2, acc[2]);
    acc[3] = wmma_bf16(af, b3, acc[3]);

    af = afn; b0 = b0n; b1 = b1n; b2 = b2n; b3 = b3n;
  }

#pragma unroll
  for (int j = 0; j < 4; ++j) {
    int ncol = n0 + j * 16 + colc;
    float bv = bias[ncol];
#pragma unroll
    for (int v = 0; v < 8; ++v) {
      int m     = row0 + v + 8 * hlf;
      float val = acc[j][v] + bv;
      if (mode == 2) {
        ((float*)out)[(size_t)m * Ndim + ncol] = val;
      } else {
        int b  = m >> 11;          // / SEQ
        int l  = m & (SEQ - 1);
        int hh = ncol >> 6;        // / DK
        int dk = ncol & (DK - 1);
        __bf16 bb = (__bf16)val;
        if (mode == 0)   // [B,H,L,DK]
          ((__bf16*)out)[((((size_t)b * HEADS + hh) * SEQ) + l) * DK + dk] = bb;
        else             // [B,H,DK,L]
          ((__bf16*)out)[((((size_t)b * HEADS + hh) * DK) + dk) * SEQ + l] = bb;
      }
    }
  }
}

// ---------------- fused flash attention ----------------
// grid: (SEQ/16, BATCH*HEADS), block: 32 (one wave)
__global__ void attention_kernel(const __bf16* __restrict__ Qb,
                                 const __bf16* __restrict__ Kb,
                                 const __bf16* __restrict__ Vt,
                                 const int* __restrict__ mask,
                                 __bf16* __restrict__ ctx) {
  int bh = blockIdx.y;
  int b  = bh >> 4;
  int hh = bh & (HEADS - 1);
  int l0 = blockIdx.x * 16;
  int lane = threadIdx.x & 31;
  int hlf  = lane >> 4;
  int col  = lane & 15;

  const __bf16* Qh = Qb + (size_t)bh * SEQ * DK;
  const __bf16* Kh = Kb + (size_t)bh * SEQ * DK;
  const __bf16* Vh = Vt + (size_t)bh * DK * SEQ;
  const int* mrow  = mask + (size_t)b * SEQ * SEQ;

  // Q A-fragments for dk 0..31 and 32..63 (loaded once)
  v16bf qa0 = load_a_frag(Qh, DK, l0, 0);
  v16bf qa1 = load_a_frag(Qh, DK, l0, 32);

  float mst[8], lst[8];
  v8f o0 = {}, o1 = {}, o2 = {}, o3 = {};
#pragma unroll
  for (int v = 0; v < 8; ++v) { mst[v] = -3.0e38f; lst[v] = 0.0f; }

  __shared__ __bf16 pls[16 * 32];
  const float scale = 0.125f;  // 1/sqrt(64)

  for (int s0 = 0; s0 < SEQ; s0 += 32) {
    // prefetch next chunk of K rows / V^T columns
    if (s0 + 32 < SEQ) {
      __builtin_prefetch(Kh + (size_t)(s0 + 32 + lane) * DK, 0, 1);
      __builtin_prefetch(Vh + (size_t)(lane << 1) * SEQ + s0 + 32, 0, 1);
    }

    // ---- front-load independent memory ops: mask + V^T fragments ----
    int ma[8], mb[8];
#pragma unroll
    for (int v = 0; v < 8; ++v) {
      int l = l0 + v + 8 * hlf;
      ma[v] = mrow[(size_t)l * SEQ + s0 + col];
      mb[v] = mrow[(size_t)l * SEQ + s0 + 16 + col];
    }
    int kb = s0 + hlf * 16;
    v16bf vf0 = *(const v16bf*)(Vh + (size_t)(0  + col) * SEQ + kb);
    v16bf vf1 = *(const v16bf*)(Vh + (size_t)(16 + col) * SEQ + kb);
    v16bf vf2 = *(const v16bf*)(Vh + (size_t)(32 + col) * SEQ + kb);
    v16bf vf3 = *(const v16bf*)(Vh + (size_t)(48 + col) * SEQ + kb);

    // ---- scores: two 16x16 tiles, K over dk=64 ----
    v8f sc0 = {}, sc1 = {};
    sc0 = wmma_bf16(qa0, load_bT_frag(Kh, DK, s0, 0), sc0);
    sc0 = wmma_bf16(qa1, load_bT_frag(Kh, DK, s0, 32), sc0);
    sc1 = wmma_bf16(qa0, load_bT_frag(Kh, DK, s0 + 16, 0), sc1);
    sc1 = wmma_bf16(qa1, load_bT_frag(Kh, DK, s0 + 16, 32), sc1);

    // ---- mask + scale ----
    float x0[8], x1[8];
#pragma unroll
    for (int v = 0; v < 8; ++v) {
      x0[v] = ma[v] ? sc0[v] * scale : -3.0e38f;
      x1[v] = mb[v] ? sc1[v] * scale : -3.0e38f;
    }

    // ---- row max over 32 cols (16 lanes per half hold the 16 cols) ----
    float t[8];
#pragma unroll
    for (int v = 0; v < 8; ++v) t[v] = fmaxf(x0[v], x1[v]);
#pragma unroll
    for (int off = 1; off < 16; off <<= 1)
#pragma unroll
      for (int v = 0; v < 8; ++v) t[v] = fmaxf(t[v], __shfl_xor(t[v], off, 32));

    float p0[8], p1[8], rs[8], corr[8];
#pragma unroll
    for (int v = 0; v < 8; ++v) {
      float mnew = fmaxf(mst[v], t[v]);
      corr[v] = __expf(mst[v] - mnew);
      mst[v]  = mnew;
      p0[v] = (x0[v] > -1.0e37f) ? __expf(x0[v] - mnew) : 0.0f;
      p1[v] = (x1[v] > -1.0e37f) ? __expf(x1[v] - mnew) : 0.0f;
      rs[v] = p0[v] + p1[v];
    }
#pragma unroll
    for (int off = 1; off < 16; off <<= 1)
#pragma unroll
      for (int v = 0; v < 8; ++v) rs[v] += __shfl_xor(rs[v], off, 32);
#pragma unroll
    for (int v = 0; v < 8; ++v) {
      lst[v] = lst[v] * corr[v] + rs[v];
      o0[v] *= corr[v]; o1[v] *= corr[v]; o2[v] *= corr[v]; o3[v] *= corr[v];
    }

    // ---- relayout P (C layout -> A fragment) through LDS ----
#pragma unroll
    for (int v = 0; v < 8; ++v) {
      int r = v + 8 * hlf;
      pls[r * 32 + col]      = (__bf16)p0[v];
      pls[r * 32 + 16 + col] = (__bf16)p1[v];
    }
    __syncthreads();
    const __bf16* pb = &pls[col * 32 + hlf * 16];
    v8bf plo = *(const v8bf*)(pb);
    v8bf phi = *(const v8bf*)(pb + 8);
    v16bf pa = __builtin_shufflevector(plo, phi,
                                       0,1,2,3,4,5,6,7,8,9,10,11,12,13,14,15);
    __syncthreads();

    // ---- O += P(16x32) * V(32x64) ----
    o0 = wmma_bf16(pa, vf0, o0);
    o1 = wmma_bf16(pa, vf1, o1);
    o2 = wmma_bf16(pa, vf2, o2);
    o3 = wmma_bf16(pa, vf3, o3);
  }

  // ---- normalize and store context bf16 [B*L, DM] ----
#pragma unroll
  for (int v = 0; v < 8; ++v) {
    float inv = (lst[v] > 0.0f) ? (1.0f / lst[v]) : 0.0f;
    int l = l0 + v + 8 * hlf;
    size_t row = ((size_t)b * SEQ + l) * DM + hh * DK;
    ctx[row + 0  + col] = (__bf16)(o0[v] * inv);
    ctx[row + 16 + col] = (__bf16)(o1[v] * inv);
    ctx[row + 32 + col] = (__bf16)(o2[v] * inv);
    ctx[row + 48 + col] = (__bf16)(o3[v] * inv);
  }
}

extern "C" void kernel_launch(void* const* d_in, const int* in_sizes, int n_in,
                              void* d_out, int out_size, void* d_ws, size_t ws_size,
                              hipStream_t stream) {
  const float* x    = (const float*)d_in[0];
  const int*   mask = (const int*)d_in[1];
  const float* Wq   = (const float*)d_in[2];
  const float* bq   = (const float*)d_in[3];
  const float* Wk   = (const float*)d_in[4];
  const float* bk   = (const float*)d_in[5];
  const float* Wv   = (const float*)d_in[6];
  const float* bv   = (const float*)d_in[7];
  const float* Wo   = (const float*)d_in[8];
  const float* bo   = (const float*)d_in[9];
  float* out = (float*)d_out;

  char* ws = (char*)d_ws;
  size_t off = 0;
  auto take = [&](size_t bytes) { char* p = ws + off; off += (bytes + 255) & ~(size_t)255; return p; };
  __bf16* xb   = (__bf16*)take((size_t)MTOT * DM * 2);
  __bf16* WqT  = (__bf16*)take((size_t)DM * DM * 2);
  __bf16* WkT  = (__bf16*)take((size_t)DM * DM * 2);
  __bf16* WvT  = (__bf16*)take((size_t)DM * DM * 2);
  __bf16* WoT  = (__bf16*)take((size_t)DM * DM * 2);
  __bf16* Qb   = (__bf16*)take((size_t)MTOT * DM * 2);
  __bf16* Kb   = (__bf16*)take((size_t)MTOT * DM * 2);
  __bf16* Vt   = (__bf16*)take((size_t)MTOT * DM * 2);
  __bf16* ctxb = (__bf16*)take((size_t)MTOT * DM * 2);

  // 1) precision casts + weight transposes
  int nX = MTOT * DM;
  cast_f32_bf16_kernel<<<nX / 256, 256, 0, stream>>>(x, xb, nX);
  dim3 tb(16, 16), tg(DM / 16, DM / 16);
  transpose_cast_kernel<<<tg, tb, 0, stream>>>(Wq, WqT, DM, DM);
  transpose_cast_kernel<<<tg, tb, 0, stream>>>(Wk, WkT, DM, DM);
  transpose_cast_kernel<<<tg, tb, 0, stream>>>(Wv, WvT, DM, DM);
  transpose_cast_kernel<<<tg, tb, 0, stream>>>(Wo, WoT, DM, DM);

  // 2) Q/K/V projections (WMMA bf16): grid = (M/16/4, N/64), 4 waves/block
  dim3 gg(MTOT / 64, DM / 64), gb(128);
  gemm_bf16_wmma_kernel<<<gg, gb, 0, stream>>>(xb, WqT, bq, Qb, MTOT, DM, DM, 0);
  gemm_bf16_wmma_kernel<<<gg, gb, 0, stream>>>(xb, WkT, bk, Kb, MTOT, DM, DM, 0);
  gemm_bf16_wmma_kernel<<<gg, gb, 0, stream>>>(xb, WvT, bv, Vt, MTOT, DM, DM, 1);

  // 3) fused flash attention
  dim3 ag(SEQ / 16, BATCH * HEADS), ab(32);
  attention_kernel<<<ag, ab, 0, stream>>>(Qb, Kb, Vt, mask, ctxb);

  // 4) output projection (fp32 out)
  gemm_bf16_wmma_kernel<<<gg, gb, 0, stream>>>(ctxb, WoT, bo, out, MTOT, DM, DM, 2);
}
